// RNN_85847806312845
// MI455X (gfx1250) — matrix operand order repaired and checked by reading, not compiled, not run on hardware
//
#include <hip/hip_runtime.h>
#include <math.h>

// ---------------------------------------------------------------------------
// PredRNN (SpatioTemporalLSTM x2 layers, 3 steps) for MI455X / gfx1250.
// Convolutions = implicit-GEMM on the WMMA pipe (v_wmma_f32_16x16x32_f16,
// f32 accumulate).  Workgroup = 128 threads (4 x wave32),
// tile = 64 Cout x 128 pixels (two image rows): each wave reuses one A
// fragment across 8 B fragments -> 8 independent back-to-back WMMAs per
// K-chunk.  Conv is templated on (CIN, KS, PAD) so im2col index math
// strength-reduces to shifts; next K-chunk's weight tile is prefetched
// (global_prefetch_b8).
// ---------------------------------------------------------------------------

typedef __attribute__((ext_vector_type(16))) _Float16 v16h;
typedef __attribute__((ext_vector_type(8)))  _Float16 v8h;
typedef __attribute__((ext_vector_type(8)))  float    v8f;

#define NHID   64
#define HW     4096      // 64*64
#define BATCH  4

// ---------------------------------------------------------------------------
// Implicit-GEMM convolution (NCHW, H=W=64).  K = CIN*KS*KS, k = tap*CIN + ci.
// grid = (128 two-row tiles [b=blk>>5, y0=(blk&31)*2], Cout/64) ; block = 128.
// Cout must be a multiple of 64 (true for all convs routed here).
// ---------------------------------------------------------------------------
template <int CIN, int KS, int PAD>
__global__ __launch_bounds__(128)
void conv_wmma(const float* __restrict__ X, const float* __restrict__ Wt,
               float* __restrict__ Y, int Cout, int inBatchStride)
{
    constexpr int KTOT = CIN * KS * KS;
    constexpr int NCH  = (KTOT + 31) >> 5;

    __shared__ _Float16 ash[64 * 32];    // A tile: [co_local][kk]          (4 KB)
    __shared__ _Float16 bsh[128 * 32];   // B tile: [n_local][kk] im2col^T  (8 KB)

    const int tid  = threadIdx.x;
    const int lane = tid & 31;
    const int wav  = tid >> 5;

    const int coBase = blockIdx.y * 64;
    const int bImg   = blockIdx.x >> 5;          // batch
    const int yRow0  = (blockIdx.x & 31) << 1;   // first of two image rows

    v8f acc0 = {}, acc1 = {}, acc2 = {}, acc3 = {};
    v8f acc4 = {}, acc5 = {}, acc6 = {}, acc7 = {};

    for (int ch = 0; ch < NCH; ++ch) {
        const int k0 = ch << 5;

        // ---- stage B: 128 pixels x 32 k (zero-fill for padding / K overrun)
        for (int e = tid; e < 4096; e += 128) {
            const int kk = e >> 7;
            const int n  = e & 127;              // 2 rows x 64 cols
            const int k  = k0 + kk;
            float v = 0.f;
            if (k < KTOT) {
                const int tap = k / CIN;         // compile-time shift/magic
                const int ci  = k - tap * CIN;
                const int ky  = tap / KS;
                const int kx  = tap - ky * KS;
                const int iy  = yRow0 + (n >> 6) + ky - PAD;
                const int ix  = (n & 63) + kx - PAD;
                if ((unsigned)iy < 64u && (unsigned)ix < 64u)
                    v = X[(size_t)bImg * inBatchStride + (size_t)ci * HW + iy * 64 + ix];
            }
            bsh[n * 32 + kk] = (_Float16)v;
        }
        // ---- stage A: 64 couts x 32 k
        for (int e = tid; e < 2048; e += 128) {
            const int co_l = e >> 5;
            const int kk   = e & 31;
            const int k    = k0 + kk;
            float v = 0.f;
            if (k < KTOT) {
                const int tap = k / CIN;
                const int ci  = k - tap * CIN;
                v = Wt[((size_t)(coBase + co_l) * CIN + ci) * (KS * KS) + tap];
            }
            ash[co_l * 32 + kk] = (_Float16)v;
        }
        // ---- prefetch next chunk's weight tile into cache (global_prefetch_b8)
        if (ch + 1 < NCH) {
            const int kn = k0 + 32 + ((tid >> 6) << 4);
            if (kn < KTOT) {
                const int tap = kn / CIN;
                const int ci  = kn - tap * CIN;
                __builtin_prefetch(
                    &Wt[((size_t)(coBase + (tid & 63)) * CIN + ci) * (KS * KS) + tap], 0, 1);
            }
        }
        __syncthreads();

        // ---- A fragment (16x32 f16, ISA 7.12.2 wave32 layout)
        // lanes 0-15: M=lane, K {0..7,16..23}; lanes 16-31: M=lane-16, K {8..15,24..31}
        const int rowA = (wav << 4) + (lane & 15);
        const int kA0  = (lane & 16) ? 8 : 0;
        const v8h a_lo = *(const v8h*)&ash[rowA * 32 + kA0];
        const v8h a_hi = *(const v8h*)&ash[rowA * 32 + kA0 + 16];
        const v16h afrag = __builtin_shufflevector(a_lo, a_hi,
            0,1,2,3,4,5,6,7,8,9,10,11,12,13,14,15);

        // ---- B fragments (32x16 each): lanes 0-15 hold K {0..7,16..23} of col N=lane,
        // lanes 16-31 hold K {8..15,24..31}.  Eight N-tiles per wave reuse one A fragment;
        // 8 disjoint accumulators issue back-to-back on the XDL pipe.
        const int nB  = lane & 15;
        const int kB0 = (lane & 16) ? 8 : 0;
#define BFRAG(j)                                                                  \
        __builtin_shufflevector(*(const v8h*)&bsh[(nB + 16 * (j)) * 32 + kB0],    \
                                *(const v8h*)&bsh[(nB + 16 * (j)) * 32 + kB0 + 16],\
                                0,1,2,3,4,5,6,7,8,9,10,11,12,13,14,15)
        acc0 = __builtin_amdgcn_wmma_f32_16x16x32_f16(false, afrag, false, BFRAG(0),
                                                      (short)0, acc0, false, false);
        acc1 = __builtin_amdgcn_wmma_f32_16x16x32_f16(false, afrag, false, BFRAG(1),
                                                      (short)0, acc1, false, false);
        acc2 = __builtin_amdgcn_wmma_f32_16x16x32_f16(false, afrag, false, BFRAG(2),
                                                      (short)0, acc2, false, false);
        acc3 = __builtin_amdgcn_wmma_f32_16x16x32_f16(false, afrag, false, BFRAG(3),
                                                      (short)0, acc3, false, false);
        acc4 = __builtin_amdgcn_wmma_f32_16x16x32_f16(false, afrag, false, BFRAG(4),
                                                      (short)0, acc4, false, false);
        acc5 = __builtin_amdgcn_wmma_f32_16x16x32_f16(false, afrag, false, BFRAG(5),
                                                      (short)0, acc5, false, false);
        acc6 = __builtin_amdgcn_wmma_f32_16x16x32_f16(false, afrag, false, BFRAG(6),
                                                      (short)0, acc6, false, false);
        acc7 = __builtin_amdgcn_wmma_f32_16x16x32_f16(false, afrag, false, BFRAG(7),
                                                      (short)0, acc7, false, false);
#undef BFRAG
        __syncthreads();
    }

    // ---- store C (VGPR r -> M=r for lanes 0-15, M=r+8 for lanes 16-31; N=lane&15)
    const int coRow0 = coBase + (wav << 4) + ((lane >> 4) << 3);
    const int xc0    = lane & 15;
    const size_t rowOff = ((size_t)bImg * Cout) << 12;
#pragma unroll
    for (int r = 0; r < 8; ++r) {
        const int co = coRow0 + r;
        float* yp0 = Y + rowOff + ((size_t)co << 12) + (yRow0 << 6);       // row y0
        float* yp1 = yp0 + 64;                                             // row y0+1
        yp0[xc0 +  0] = acc0[r];
        yp0[xc0 + 16] = acc1[r];
        yp0[xc0 + 32] = acc2[r];
        yp0[xc0 + 48] = acc3[r];
        yp1[xc0 +  0] = acc4[r];
        yp1[xc0 + 16] = acc5[r];
        yp1[xc0 + 32] = acc6[r];
        yp1[xc0 + 48] = acc7[r];
    }
}

// ---------------------------------------------------------------------------
// Elementwise gate kernels
// ---------------------------------------------------------------------------
__device__ __forceinline__ float sigmoidf_(float x) { return 1.f / (1.f + __expf(-x)); }

// c_new = f*c + i*g ; m_new = f'*m + i'*g' ; memcat = [c_new ; m_new]
__global__ void gate1_kernel(const float* __restrict__ xc, const float* __restrict__ hc,
                             const float* __restrict__ mc,
                             float* __restrict__ c, float* __restrict__ m,
                             float* __restrict__ memcat)
{
    const int idx = blockIdx.x * blockDim.x + threadIdx.x;
    if (idx >= BATCH * NHID * HW) return;
    const int pix = idx & 4095;
    const int cch = (idx >> 12) & 63;
    const int b   = idx >> 18;

#define XI(g) ((((size_t)(b * 7 * NHID + (g) * NHID + cch)) << 12) + pix)
#define HI(g) ((((size_t)(b * 4 * NHID + (g) * NHID + cch)) << 12) + pix)
#define MI(g) ((((size_t)(b * 3 * NHID + (g) * NHID + cch)) << 12) + pix)
    const size_t si = (((size_t)(b * NHID + cch)) << 12) + pix;

    const float it = sigmoidf_(xc[XI(0)] + hc[HI(0)]);
    const float ft = sigmoidf_(xc[XI(1)] + hc[HI(1)] + 1.0f);
    const float gt = tanhf    (xc[XI(2)] + hc[HI(2)]);
    const float cn = ft * c[si] + it * gt;

    const float itp = sigmoidf_(xc[XI(3)] + mc[MI(0)]);
    const float ftp = sigmoidf_(xc[XI(4)] + mc[MI(1)] + 1.0f);
    const float gtp = tanhf    (xc[XI(5)] + mc[MI(2)]);
    const float mn  = ftp * m[si] + itp * gtp;

    c[si] = cn;
    m[si] = mn;
    memcat[(((size_t)(b * 2 * NHID + cch)) << 12) + pix]        = cn;
    memcat[(((size_t)(b * 2 * NHID + NHID + cch)) << 12) + pix] = mn;
}

// h = sigmoid(o_x + o_h + conv_o) * tanh(conv_c)
__global__ void gate2_kernel(const float* __restrict__ xc, const float* __restrict__ hc,
                             const float* __restrict__ ocv, const float* __restrict__ ccv,
                             float* __restrict__ h)
{
    const int idx = blockIdx.x * blockDim.x + threadIdx.x;
    if (idx >= BATCH * NHID * HW) return;
    const int pix = idx & 4095;
    const int cch = (idx >> 12) & 63;
    const int b   = idx >> 18;
    const size_t si = (((size_t)(b * NHID + cch)) << 12) + pix;
    const float ot = sigmoidf_(xc[XI(6)] + hc[HI(3)] + ocv[si]);
    h[si] = ot * tanhf(ccv[si]);
}
#undef XI
#undef HI
#undef MI

// x_gen = 1x1 conv (NHID -> 1) with Wlast; optionally also writes output slice
__global__ void xgen_kernel(const float* __restrict__ h1, const float* __restrict__ Wlast,
                            float* __restrict__ xg, float* __restrict__ out_slice)
{
    const int idx = blockIdx.x * blockDim.x + threadIdx.x;
    if (idx >= BATCH * HW) return;
    const int b = idx >> 12, pix = idx & 4095;
    float s = 0.f;
#pragma unroll 8
    for (int ci = 0; ci < NHID; ++ci)
        s += Wlast[ci] * h1[(((size_t)(b * NHID + ci)) << 12) + pix];
    xg[idx] = s;
    if (out_slice) out_slice[(size_t)b * (2 * HW) + pix] = s;
}

// ---------------------------------------------------------------------------
// Host orchestration (graph-capture safe: only async launches / memsets)
// ---------------------------------------------------------------------------
extern "C" void kernel_launch(void* const* d_in, const int* in_sizes, int n_in,
                              void* d_out, int out_size, void* d_ws, size_t ws_size,
                              hipStream_t stream)
{
    (void)in_sizes; (void)n_in; (void)out_size; (void)ws_size;

    const float* frames = (const float*)d_in[0];
    const float* Wx0 = (const float*)d_in[1];
    const float* Wh0 = (const float*)d_in[2];
    const float* Wm0 = (const float*)d_in[3];
    const float* Wo0 = (const float*)d_in[4];
    const float* Wc0 = (const float*)d_in[5];
    const float* Wx1 = (const float*)d_in[6];
    const float* Wh1 = (const float*)d_in[7];
    const float* Wm1 = (const float*)d_in[8];
    const float* Wo1 = (const float*)d_in[9];
    const float* Wc1 = (const float*)d_in[10];
    const float* Wlast = (const float*)d_in[11];
    float* out = (float*)d_out;

    float* ws = (float*)d_ws;
    size_t off = 0;
    auto alloc = [&](size_t n) { float* p = ws + off; off += n; return p; };

    const size_t SZ = (size_t)BATCH * NHID * HW;     // one [B,NH,H,W] tensor
    float* xc     = alloc((size_t)BATCH * 7 * NHID * HW);
    float* hcb    = alloc((size_t)BATCH * 4 * NHID * HW);
    float* mcb    = alloc((size_t)BATCH * 3 * NHID * HW);
    float* ocv    = alloc(SZ);
    float* ccv    = alloc(SZ);
    float* memcat = alloc(2 * SZ);
    float* h0     = alloc(SZ);           // h0,c0,h1,c1,mem contiguous -> single memset
    float* c0     = alloc(SZ);
    float* h1     = alloc(SZ);
    float* c1     = alloc(SZ);
    float* mem    = alloc(SZ);
    float* xg     = alloc((size_t)BATCH * HW);

    hipMemsetAsync(h0, 0, 5 * SZ * sizeof(float), stream);   // zero recurrent state

    const dim3 blk(128);
    auto conv = [&](const float* X, const float* W, float* Y,
                    int Cin, int Cout, int KS, int ibs) {
        dim3 grd(128, Cout / 64);
        if (Cin == 1 && KS == 5)
            conv_wmma<1, 5, 2><<<grd, blk, 0, stream>>>(X, W, Y, Cout, ibs);
        else if (Cin == 64 && KS == 5)
            conv_wmma<64, 5, 2><<<grd, blk, 0, stream>>>(X, W, Y, Cout, ibs);
        else if (Cin == 128 && KS == 5)
            conv_wmma<128, 5, 2><<<grd, blk, 0, stream>>>(X, W, Y, Cout, ibs);
        else // Cin == 128, KS == 1
            conv_wmma<128, 1, 0><<<grd, blk, 0, stream>>>(X, W, Y, Cout, ibs);
    };

    const int ELT = BATCH * NHID * HW;   // 1,048,576
    for (int t = 0; t < 3; ++t) {
        // ----- layer 0 -----
        const float* net;
        int ibs;
        if (t < 2) { net = frames + (size_t)t * HW; ibs = 4 * HW; }  // frames[:, t]
        else       { net = xg;                      ibs = HW;     }  // x_gen feedback
        conv(net,    Wx0, xc,  1,        7 * NHID, 5, ibs);
        conv(h0,     Wh0, hcb, NHID,     4 * NHID, 5, NHID * HW);
        conv(mem,    Wm0, mcb, NHID,     3 * NHID, 5, NHID * HW);
        gate1_kernel<<<ELT / 256, 256, 0, stream>>>(xc, hcb, mcb, c0, mem, memcat);
        conv(memcat, Wo0, ocv, 2 * NHID, NHID,     5, 2 * NHID * HW);
        conv(memcat, Wc0, ccv, 2 * NHID, NHID,     1, 2 * NHID * HW);
        gate2_kernel<<<ELT / 256, 256, 0, stream>>>(xc, hcb, ocv, ccv, h0);

        // ----- layer 1 -----
        conv(h0,     Wx1, xc,  NHID,     7 * NHID, 5, NHID * HW);
        conv(h1,     Wh1, hcb, NHID,     4 * NHID, 5, NHID * HW);
        conv(mem,    Wm1, mcb, NHID,     3 * NHID, 5, NHID * HW);
        gate1_kernel<<<ELT / 256, 256, 0, stream>>>(xc, hcb, mcb, c1, mem, memcat);
        conv(memcat, Wo1, ocv, 2 * NHID, NHID,     5, 2 * NHID * HW);
        conv(memcat, Wc1, ccv, 2 * NHID, NHID,     1, 2 * NHID * HW);
        gate2_kernel<<<ELT / 256, 256, 0, stream>>>(xc, hcb, ocv, ccv, h1);

        // ----- x_gen ; last 2 timesteps feed the output [B,2,1,64,64] -----
        float* oslice = (t >= 1) ? (out + (size_t)(t - 1) * HW) : nullptr;
        xgen_kernel<<<(BATCH * HW) / 256, 256, 0, stream>>>(h1, Wlast, xg, oslice);
    }
}
